// EncoderLayer_28595892256996
// MI455X (gfx1250) — matrix operand order repaired and checked by reading, not compiled, not run on hardware
//
#include <hip/hip_runtime.h>
#include <hip/hip_bf16.h>

typedef __attribute__((ext_vector_type(16))) _Float16 v16h;
typedef __attribute__((ext_vector_type(8)))  _Float16 v8h;
typedef __attribute__((ext_vector_type(4)))  _Float16 v4h;
typedef __attribute__((ext_vector_type(8)))  float    v8f;
typedef __attribute__((ext_vector_type(4)))  float    v4f;

typedef unsigned int u32;
typedef __attribute__((ext_vector_type(4))) u32 u32x4;
typedef __attribute__((ext_vector_type(8))) int i32x8;
typedef __attribute__((ext_vector_type(4))) int i32x4;

#define DIM   128
#define NROW  262144      // N
#define MROW  131072      // M (power of two: r>>17 = b, r&(M-1) = m)
#define BQ    4
#define K1    768         // 3 * 2*DIM   (vec folded into A)
#define K2    256         // 2 * DIM
#define ROWS_PER_WAVE   16
#define WAVES_PER_BLOCK 4
#define ROWS_PER_BLOCK  64

// ---------------------------------------------------------------------------
// TDM: DMA a 2D tile (tile_d0 x tile_d1 f16 elements, row stride stride0
// elements) from global into LDS at lds_off, packed row-major (tile_d0*2 bytes
// per row, rows contiguous).  D# layout per CDNA5 ISA ch.8 (§8.3/§8.4).
// ---------------------------------------------------------------------------
__device__ __forceinline__ void tdm_load_2d_f16(u32 lds_off, const void* gptr,
                                                u32 tile_d0, u32 tile_d1,
                                                u32 stride0) {
  unsigned long long ga = (unsigned long long)(size_t)gptr;
  u32x4 g0;
  g0[0] = 1u;                                             // count=1, user mode
  g0[1] = lds_off;                                        // lds_addr [63:32]
  g0[2] = (u32)(ga & 0xffffffffu);                        // global_addr lo
  g0[3] = (u32)((ga >> 32) & 0x01ffffffu) | (2u << 30);   // addr hi | type=2
  i32x8 g1;
  g1[0] = (int)(1u << 16);                                // data_size = 2 bytes
  g1[1] = (int)((tile_d0 & 0xffffu) << 16);               // tensor_dim0 lo16
  g1[2] = (int)(((tile_d0 >> 16) & 0xffffu) |             // tensor_dim0 hi16
                ((tile_d1 & 0xffffu) << 16));             // tensor_dim1 lo16
  g1[3] = (int)(((tile_d1 >> 16) & 0xffffu) |             // tensor_dim1 hi16
                ((tile_d0 & 0xffffu) << 16));             // tile_dim0
  g1[4] = (int)(tile_d1 & 0xffffu);                       // tile_dim1 (dim2=0)
  g1[5] = (int)stride0;                                   // dim0_stride lo32
  g1[6] = 0;
  g1[7] = 0;
  i32x4 z4 = {0, 0, 0, 0};
#if defined(__clang_major__) && __clang_major__ >= 23
  i32x8 z8 = {0, 0, 0, 0, 0, 0, 0, 0};
  __builtin_amdgcn_tensor_load_to_lds(g0, g1, z4, z4, z8, 0);
#else
  __builtin_amdgcn_tensor_load_to_lds(g0, g1, z4, z4, 0);
#endif
}

// ---------------------------------------------------------------------------
// Weight prep: W_bil [128,256,3] f32 -> Wcat [128,768] f16  (k = j*256 + i)
//              W_ms  [128,256]   f32 -> Wms16 [128,256] f16
// ---------------------------------------------------------------------------
__global__ void prep_weights(const float* __restrict__ W_bil,
                             const float* __restrict__ W_ms,
                             _Float16* __restrict__ Wcat,
                             _Float16* __restrict__ Wms16) {
  int idx = blockIdx.x * blockDim.x + threadIdx.x;
  if (idx < DIM * K1) {
    int o = idx / K1;
    int k = idx - o * K1;
    int j = k >> 8;        // k / 256
    int i = k & 255;       // k % 256
    Wcat[idx] = (_Float16)W_bil[(o * K2 + i) * 3 + j];
  } else {
    int t = idx - DIM * K1;
    if (t < DIM * K2) Wms16[t] = (_Float16)W_ms[t];
  }
}

// ---------------------------------------------------------------------------
// Fused encoder.  One wave = 16 rows x 128 cols; 4 waves/block.
// B-matrix slices (32 K x 128 o, 8KB f16) are TDM-DMA'd into a double-
// buffered LDS stage, overlapped with the WMMAs of the previous slice.
// TDM packs the slice as [o][32 halves], which this code reads directly as its B-fragment
// layout: col o = t*16 + (lane&15), 16 contiguous K halves per lane-half.
// ---------------------------------------------------------------------------
__launch_bounds__(128)
__global__ void encoder_fused(const float* __restrict__ last,
                              const float* __restrict__ sample,
                              const float* __restrict__ vec,
                              const int*   __restrict__ cl,
                              const int*   __restrict__ cr,
                              const int*   __restrict__ cs,
                              const float* __restrict__ b_bil,
                              const float* __restrict__ b_ms,
                              const float* __restrict__ prelu_a,
                              const float* __restrict__ drop,
                              const _Float16* __restrict__ Wcat,
                              const _Float16* __restrict__ Wms16,
                              float* __restrict__ out) {
  // Per-wave 16x256 f16 tile: first x = [last_l | last_r], reused as
  // [h | sample_s] for stage B.  4 waves * 8KB = 32KB.
  __shared__ __align__(32) _Float16 xtile[WAVES_PER_BLOCK][ROWS_PER_WAVE][K2];
  // Double-buffered TDM landing zone for B slices: 2 * 128 rows * 64B = 16KB.
  __shared__ __align__(64) _Float16 bbuf[2][DIM][32];

  const int tid   = threadIdx.x;
  const int wave  = tid >> 5;
  const int lane  = tid & 31;
  const int rhalf = lane >> 4;    // K/row half this lane serves
  const int rlane = lane & 15;    // row within wave tile (A) / col within tile (B,C)

  const long rowBase = (long)blockIdx.x * ROWS_PER_BLOCK + (long)wave * ROWS_PER_WAVE;
  const long rMine   = rowBase + rlane;
  const int  bIdx    = (int)(rMine >> 17);
  const int  mIdx    = (int)(rMine & (MROW - 1));

  _Float16* myLds = &xtile[wave][0][0];
  const u32 bbufOff0 = (u32)(size_t)(void*)&bbuf[0][0][0];
  const u32 bbufOff1 = (u32)(size_t)(void*)&bbuf[1][0][0];

  // ---- stage x = concat(last[b,cl[m]], last[b,cr[m]]) as f16 into LDS ----
  {
    const int child   = rhalf ? cr[mIdx] : cl[mIdx];
    const float* src  = last + ((long)bIdx * NROW + (long)child) * DIM;
    _Float16*    dst  = myLds + rlane * K2 + rhalf * DIM;
#pragma unroll
    for (int c = 0; c < DIM; c += 4) {
      v4f f = *(const v4f*)(src + c);
      v4h h = { (_Float16)f[0], (_Float16)f[1], (_Float16)f[2], (_Float16)f[3] };
      *(v4h*)(dst + c) = h;
    }
  }

  // per-row vec scalars (folded into A fragments of stage A)
  const float* vp = vec + rMine * 3;
  const _Float16 v0h = (_Float16)vp[0];
  const _Float16 v1h = (_Float16)vp[1];
  const _Float16 v2h = (_Float16)vp[2];

  const int khalf = rhalf * 16;   // A-frag K sub-offset within a 32-wide K block

  // ================= Stage A: h_pre = A''(16x768) @ Wcat^T =================
  v8f acc[8] = {};
  if (wave == 0)
    tdm_load_2d_f16(bbufOff0, Wcat, 32, DIM, K1);   // slice ks=0 -> buf 0

  int cur = 0;
#pragma unroll 1
  for (int ks = 0; ks < 24; ++ks) {
    if (wave == 0) __builtin_amdgcn_s_wait_tensorcnt(0);
    __syncthreads();                                // slice `cur` published
    if (wave == 0 && ks + 1 < 24)
      tdm_load_2d_f16(cur ? bbufOff0 : bbufOff1,    // prefetch next slice
                      Wcat + (ks + 1) * 32, 32, DIM, K1);

    // A fragment: lane row = rlane, K = {khalf..khalf+15}
    const int j    = ks >> 3;                       // which vec component
    const _Float16 vjs = (j == 0) ? v0h : ((j == 1) ? v1h : v2h);
    const int xoff = (ks & 7) * 32;                 // K pos inside 256-wide x
    const _Float16* arow = myLds + rlane * K2 + xoff;
    v8h lo = *(const v8h*)(arow + khalf);
    v8h hi = *(const v8h*)(arow + 8 + khalf);
    v16h afrag;
#pragma unroll
    for (int e = 0; e < 8; ++e) {
      afrag[e]     = lo[e] * vjs;
      afrag[8 + e] = hi[e] * vjs;
    }

    // B fragments straight out of the TDM-packed LDS slice
    const _Float16* bslice = &bbuf[cur][0][0];
    v16h bf[8];
#pragma unroll
    for (int t = 0; t < 8; ++t)
      bf[t] = *(const v16h*)(bslice + (t * 16 + rlane) * 32 + rhalf * 16);
#pragma unroll
    for (int t = 0; t < 8; ++t)
      acc[t] = __builtin_amdgcn_wmma_f32_16x16x32_f16(
          false, afrag, false, bf[t], (short)0, acc[t], false, false);

    cur ^= 1;
  }

  const float a_slope = prelu_a[0];

  // ---- epilogue A: bias + PReLU, f16 -> own LDS tile cols [0,128) ----
  // C/D layout: lane L element g -> row M = g + 8*(L>>4), col N = L&15
#pragma unroll
  for (int t = 0; t < 8; ++t) {
    const int   o  = t * 16 + rlane;
    const float bb = b_bil[o];
#pragma unroll
    for (int g = 0; g < 8; ++g) {
      const int rloc = g + rhalf * 8;
      float h = acc[t][g] + bb;
      h = (h >= 0.0f) ? h : a_slope * h;
      myLds[rloc * K2 + o] = (_Float16)h;
    }
  }

  // ---- stage s = sample[b, cs[m]] as f16 into own tile cols [128,256) ----
  {
    const int child  = cs[mIdx];
    const float* src = sample + ((long)bIdx * NROW + (long)child) * DIM + rhalf * 64;
    _Float16*    dst = myLds + rlane * K2 + DIM + rhalf * 64;
#pragma unroll
    for (int c = 0; c < 64; c += 4) {
      v4f f = *(const v4f*)(src + c);
      v4h h = { (_Float16)f[0], (_Float16)f[1], (_Float16)f[2], (_Float16)f[3] };
      *(v4h*)(dst + c) = h;
    }
  }

  // ================= Stage B: y_pre = [h|s](16x256) @ Wms^T ================
  __syncthreads();          // all waves done with bbuf from stage A
  v8f acc2[8] = {};
  if (wave == 0)
    tdm_load_2d_f16(bbufOff0, Wms16, 32, DIM, K2);  // slice ks=0 -> buf 0

  cur = 0;
#pragma unroll 1
  for (int ks = 0; ks < 8; ++ks) {
    if (wave == 0) __builtin_amdgcn_s_wait_tensorcnt(0);
    __syncthreads();
    if (wave == 0 && ks + 1 < 8)
      tdm_load_2d_f16(cur ? bbufOff0 : bbufOff1,
                      Wms16 + (ks + 1) * 32, 32, DIM, K2);

    const _Float16* arow = myLds + rlane * K2 + ks * 32;
    v8h lo = *(const v8h*)(arow + khalf);
    v8h hi = *(const v8h*)(arow + 8 + khalf);
    v16h afrag;
#pragma unroll
    for (int e = 0; e < 8; ++e) {
      afrag[e]     = lo[e];
      afrag[8 + e] = hi[e];
    }

    const _Float16* bslice = &bbuf[cur][0][0];
    v16h bf[8];
#pragma unroll
    for (int t = 0; t < 8; ++t)
      bf[t] = *(const v16h*)(bslice + (t * 16 + rlane) * 32 + rhalf * 16);
#pragma unroll
    for (int t = 0; t < 8; ++t)
      acc2[t] = __builtin_amdgcn_wmma_f32_16x16x32_f16(
          false, afrag, false, bf[t], (short)0, acc2[t], false, false);

    cur ^= 1;
  }

  // ---- epilogue B: bias + PReLU + dropout mask, f32 out ----
#pragma unroll
  for (int t = 0; t < 8; ++t) {
    const int   o  = t * 16 + rlane;
    const float bb = b_ms[o];
#pragma unroll
    for (int g = 0; g < 8; ++g) {
      const long rout = rowBase + g + rhalf * 8;
      float y = acc2[t][g] + bb;
      y = (y >= 0.0f) ? y : a_slope * y;
      y *= drop[rout * DIM + o];
      out[rout * DIM + o] = y;
    }
  }
}

// ---------------------------------------------------------------------------
extern "C" void kernel_launch(void* const* d_in, const int* in_sizes, int n_in,
                              void* d_out, int out_size, void* d_ws, size_t ws_size,
                              hipStream_t stream) {
  (void)in_sizes; (void)n_in; (void)out_size; (void)ws_size;

  const float* last    = (const float*)d_in[0];
  const float* sample  = (const float*)d_in[1];
  const float* vec     = (const float*)d_in[2];
  const int*   cl      = (const int*)d_in[3];
  const int*   cr      = (const int*)d_in[4];
  const int*   cs      = (const int*)d_in[5];
  const float* W_bil   = (const float*)d_in[6];
  const float* b_bil   = (const float*)d_in[7];
  const float* W_ms    = (const float*)d_in[8];
  const float* b_ms    = (const float*)d_in[9];
  const float* prelu_a = (const float*)d_in[10];
  const float* drop    = (const float*)d_in[11];

  _Float16* Wcat  = (_Float16*)d_ws;              // 128*768 f16 = 192KB
  _Float16* Wms16 = Wcat + (size_t)DIM * K1;      // 128*256 f16 =  64KB

  const int prepThreads = DIM * K1 + DIM * K2;    // 131072
  prep_weights<<<(prepThreads + 255) / 256, 256, 0, stream>>>(W_bil, W_ms, Wcat, Wms16);

  const long totalRows = (long)BQ * MROW;         // 524288
  const int  grid      = (int)(totalRows / ROWS_PER_BLOCK);  // 8192 blocks
  encoder_fused<<<grid, 128, 0, stream>>>(last, sample, vec, cl, cr, cs,
                                          b_bil, b_ms, prelu_a, drop,
                                          Wcat, Wms16, (float*)d_out);
}
